// ContrastiveLoss_8306466750807
// MI455X (gfx1250) — matrix operand order repaired and checked by reading, not compiled, not run on hardware
//
#include <hip/hip_runtime.h>
#include <hip/hip_bf16.h>
#include <math.h>

// ---------------- problem constants ----------------
#define NN        4096               // rows per embedding
#define DIM       512                // embedding dim
#define KSTEPS    (DIM / 32)         // 16 WMMA k-steps per 16x16 tile
#define MM        (2 * NN)           // 8192 total rows
#define WAVES_WG  8
#define THREADS_WG (WAVES_WG * 32)   // 256
#define ROWS_WG   (WAVES_WG * 16)    // 128 rows per workgroup
#define COL_SLABS 16
#define COLS_SLAB (MM / COL_SLABS)   // 512 columns per slab
#define TILE_HALF (16 * DIM)         // halves in one 16-column B panel (16 KB)
#define CHUNKS_PER_THREAD (TILE_HALF / (THREADS_WG * 8))   // 4 x b128 per thread
#define INV_T     2.0f               // 1 / TEMPERATURE

typedef _Float16 v8h  __attribute__((ext_vector_type(8)));
typedef _Float16 v16h __attribute__((ext_vector_type(16)));
typedef float    v8f  __attribute__((ext_vector_type(8)));

union V16 { v16h v; v8h h[2]; };

// ---------------------------------------------------------------------------
// Async copy of one contiguous 16 KB B panel (16 cols x 512 halves) into LDS.
// CDNA5 GLOBAL_LOAD_ASYNC_TO_LDS_B128: VDST = LDS byte offset (addr[31:0] of
// the flat address), VADDR = 64-bit global address. Tracked by ASYNCcnt.
// ---------------------------------------------------------------------------
__device__ __forceinline__ void stage_tile_async(const _Float16* __restrict__ g,
                                                 _Float16* l, int tid) {
#pragma unroll
    for (int i = 0; i < CHUNKS_PER_THREAD; ++i) {
        const int off = (tid + THREADS_WG * i) * 8;              // halves (16B chunks)
        const unsigned lofs = (unsigned)(uintptr_t)(l + off);    // LDS byte offset
        asm volatile("global_load_async_to_lds_b128 %0, %1, off"
                     :: "v"(lofs), "v"(g + off)
                     : "memory");
    }
}

__device__ __forceinline__ void wait_async_and_barrier() {
    asm volatile("s_wait_asynccnt 0x0" ::: "memory");
    __syncthreads();
}

// ---------------------------------------------------------------------------
// Kernel 1: L2-normalize each row in f32, emit f16 reps[MM][DIM].
// ---------------------------------------------------------------------------
__global__ __launch_bounds__(256) void nk_normalize(const float* __restrict__ embi,
                                                    const float* __restrict__ embj,
                                                    _Float16* __restrict__ reps) {
    const int tid  = threadIdx.x;
    const int wave = tid >> 5;
    const int lane = tid & 31;
    const int row  = blockIdx.x * 8 + wave;          // 0..8191
    const float* src = (row < NN) ? (embi + (size_t)row * DIM)
                                  : (embj + (size_t)(row - NN) * DIM);
    float x[16];
    float ss = 0.0f;
#pragma unroll
    for (int t = 0; t < 16; ++t) {
        x[t] = src[lane + 32 * t];
        ss += x[t] * x[t];
    }
#pragma unroll
    for (int mask = 1; mask <= 16; mask <<= 1)
        ss += __shfl_xor(ss, mask, 32);
    const float inv = 1.0f / sqrtf(ss);
    _Float16* dst = reps + (size_t)row * DIM;
#pragma unroll
    for (int t = 0; t < 16; ++t)
        dst[lane + 32 * t] = (_Float16)(x[t] * inv);
}

// ---------------------------------------------------------------------------
// Kernel 2: fused sim-GEMM + online exp row-reduction.
//   - A strip (16 rows x 512) register-resident per wave (128 VGPRs).
//   - B panels double-buffered in LDS via async-to-LDS DMA (ASYNCcnt),
//     prefetched one full tile ahead; shared by all 8 waves (8x reuse).
//   - B fragments read with ds_load_b128; 16x v_wmma_f32_16x16x32_f16 per tile.
// ---------------------------------------------------------------------------
__global__ __launch_bounds__(THREADS_WG) void nk_simreduce(const _Float16* __restrict__ reps,
                                                           float* __restrict__ denomPart,
                                                           float* __restrict__ posPart) {
    __shared__ __align__(16) _Float16 ldsB[2][TILE_HALF];   // 2 x 16 KB

    const int tid     = threadIdx.x;
    const int wave    = tid >> 5;
    const int lane    = tid & 31;
    const int lanelo  = lane & 15;
    const int halfsel = lane >> 4;                 // 0: lanes 0-15, 1: lanes 16-31
    const int r0      = blockIdx.x * ROWS_WG;      // global first row of this WG
    const int slab    = blockIdx.y;
    const int jbase   = slab * COLS_SLAB;

    // ---- A fragments -> registers, straight from global (one-time) ----
    // lane holds row (wave*16 + lanelo):
    //   halves 0-7  = K + 8*halfsel + {0..7}; halves 8-15 = K + 16 + 8*halfsel + {0..7}
    const _Float16* Ab = reps + (size_t)(r0 + wave * 16 + lanelo) * DIM + 8 * halfsel;
    V16 a[KSTEPS];
#pragma unroll
    for (int kt = 0; kt < KSTEPS; ++kt) {
        a[kt].h[0] = *(const v8h*)(Ab + 32 * kt);
        a[kt].h[1] = *(const v8h*)(Ab + 32 * kt + 16);
    }

    // C-tile row mapping: VGPR r -> row rowBaseC + r, column = j0 + lanelo
    const int rowBaseC = r0 + wave * 16 + halfsel * 8;

    float denom_acc[8], pos_acc[8];
#pragma unroll
    for (int r = 0; r < 8; ++r) { denom_acc[r] = 0.0f; pos_acc[r] = 0.0f; }

    // ---- prime: async-stage tile 0 into buffer 0 ----
    stage_tile_async(reps + (size_t)jbase * DIM, &ldsB[0][0], tid);
    wait_async_and_barrier();

    int p = 0;
    for (int jt = 0; jt < COLS_SLAB; jt += 16) {
        // async-prefetch next B panel into the other buffer (a full tile ahead)
        if (jt + 16 < COLS_SLAB)
            stage_tile_async(reps + (size_t)(jbase + jt + 16) * DIM, &ldsB[p ^ 1][0], tid);

        // B fragment base in LDS: lane reads "column" lanelo, halves
        // K + 8*halfsel + {0..7} and K + 16 + 8*halfsel + {0..7} per k-step (two ds_load_b128)
        const _Float16* Bb = &ldsB[p][0] + (size_t)lanelo * DIM + 8 * halfsel;

        v8f c = {};
#pragma unroll
        for (int kt = 0; kt < KSTEPS; ++kt) {
            V16 b;
            b.h[0] = *(const v8h*)(Bb + 32 * kt);       // ds_load_b128
            b.h[1] = *(const v8h*)(Bb + 32 * kt + 16);  // ds_load_b128
            c = __builtin_amdgcn_wmma_f32_16x16x32_f16(
                    /*neg_a=*/false, a[kt].v, /*neg_b=*/false, b.v,
                    /*c_mod=*/(short)0, c, /*reuse_a=*/false, /*reuse_b=*/false);
        }

        // online reduction of this 16x16 sim tile
        const int colj = jbase + jt + lanelo;
#pragma unroll
        for (int r = 0; r < 8; ++r) {
            const int rowi = rowBaseC + r;
            const float sim = INV_T * c[r];
            const float e = __expf(sim);
            denom_acc[r] += (colj == rowi) ? 0.0f : e;     // drop diagonal
            const int partner = (rowi < NN) ? rowi + NN : rowi - NN;
            pos_acc[r] += (colj == partner) ? sim : 0.0f;  // at most one hit
        }

        wait_async_and_barrier();   // next panel landed; prev buffer free
        p ^= 1;
    }

    // reduce across the 16 lanes of each half (columns of the tile)
#pragma unroll
    for (int r = 0; r < 8; ++r) {
        for (int mask = 1; mask <= 8; mask <<= 1) {
            denom_acc[r] += __shfl_xor(denom_acc[r], mask, 32);
            pos_acc[r]   += __shfl_xor(pos_acc[r], mask, 32);
        }
    }
    if (lanelo == 0) {
        float* dslab = denomPart + (size_t)slab * MM;
        float* pslab = posPart   + (size_t)slab * MM;
#pragma unroll
        for (int r = 0; r < 8; ++r) {
            const int rowi = rowBaseC + r;
            dslab[rowi] = denom_acc[r];   // written exactly once per launch
            pslab[rowi] = pos_acc[r];
        }
    }
}

// ---------------------------------------------------------------------------
// Kernel 3: loss = mean_i( log(sum_slab denom) - sum_slab pos )
// ---------------------------------------------------------------------------
__global__ __launch_bounds__(1024) void nk_loss(const float* __restrict__ denomPart,
                                                const float* __restrict__ posPart,
                                                float* __restrict__ out) {
    __shared__ float red[1024];
    float s = 0.0f;
    for (int i = threadIdx.x; i < MM; i += 1024) {
        float denom = 0.0f, pos = 0.0f;
#pragma unroll
        for (int sl = 0; sl < COL_SLABS; ++sl) {
            denom += denomPart[(size_t)sl * MM + i];
            pos   += posPart[(size_t)sl * MM + i];
        }
        s += logf(denom) - pos;
    }
    red[threadIdx.x] = s;
    __syncthreads();
    for (int off = 512; off > 0; off >>= 1) {
        if (threadIdx.x < off) red[threadIdx.x] += red[threadIdx.x + off];
        __syncthreads();
    }
    if (threadIdx.x == 0) out[0] = red[0] / (float)MM;
}

// ---------------------------------------------------------------------------
extern "C" void kernel_launch(void* const* d_in, const int* in_sizes, int n_in,
                              void* d_out, int out_size, void* d_ws, size_t ws_size,
                              hipStream_t stream) {
    const float* embi = (const float*)d_in[0];
    const float* embj = (const float*)d_in[1];
    float* out = (float*)d_out;

    // workspace layout:
    //   [0, 8MB)         : f16 reps[MM][DIM]
    //   [+0, +512KB)     : denomPart[COL_SLABS][MM] f32
    //   [+512KB, +1MB)   : posPart  [COL_SLABS][MM] f32
    char* ws = (char*)d_ws;
    _Float16* reps   = (_Float16*)ws;
    float* denomPart = (float*)(ws + (size_t)MM * DIM * sizeof(_Float16));
    float* posPart   = denomPart + (size_t)COL_SLABS * MM;

    nk_normalize<<<MM / 8, 256, 0, stream>>>(embi, embj, reps);

    dim3 grid(MM / ROWS_WG, COL_SLABS);   // 64 x 16 = 1024 workgroups
    nk_simreduce<<<grid, THREADS_WG, 0, stream>>>(reps, denomPart, posPart);

    nk_loss<<<1, 1024, 0, stream>>>(denomPart, posPart, out);
}
